// IndexTTS_A_65206193488313
// MI455X (gfx1250) — matrix operand order repaired and checked by reading, not compiled, not run on hardware
//
#include <hip/hip_runtime.h>
#include <hip/hip_bf16.h>
#include <math.h>

// ---------------------------------------------------------------------------
// Types for CDNA5 WMMA (wave32, v_wmma_f32_16x16x32_bf16)
// ---------------------------------------------------------------------------
typedef __bf16 bf16;
typedef __attribute__((ext_vector_type(16))) bf16  v16bf;
typedef __attribute__((ext_vector_type(8)))  bf16  v8bf;
typedef __attribute__((ext_vector_type(8)))  float v8f;

enum { T_ = 2048, D_ = 512, H_ = 8, DH_ = 64, NL_ = 4, FF_ = 2048, KC_ = 31,
       NLAT_ = 32, NP_ = 2, PH_ = 8, PDH_ = 64, S_ = T_ + NLAT_ };

__device__ __forceinline__ bf16 f2bf(float f) {
    unsigned u = __float_as_uint(f);
    unsigned r = (u + 0x7FFFu + ((u >> 16) & 1u)) >> 16;   // round-to-nearest-even
    unsigned short hs = (unsigned short)r;
    return __builtin_bit_cast(bf16, hs);
}
__device__ __forceinline__ int imin(int a, int b) { return a < b ? a : b; }

// ---------------------------------------------------------------------------
// WMMA GEMM: C[M,N] = A[M,K] (row-major, lda) * B[N,K]^T (row-major, ldb) + bias[N]
// Block: 256 threads = 8 waves. Wave tile: 32 x (16*JT). Block tile: 128 x (32*JT).
// JT=2 (N<128 GEMMs) -> 4 WMMA/k-step; JT=4 (large-N GEMMs) -> 8 WMMA/k-step,
// 1.5 b128 loads per WMMA. K%32==0 for all call sites.
// Fragment layouts per CDNA5 ISA 7.12.2:
//   A (16x32 bf16): lane row=l&15, kb=(l>>4)*8, elems K=kb..kb+7, 16+kb..16+kb+7
//   B (32x16 bf16): lane col=l&15, kb=(l>>4)*8, elems K=kb..kb+7, 16+kb..16+kb+7
//   C (16x16 f32) : lane col=l&15, VGPR e -> row e + 8*(l>>4)
// ---------------------------------------------------------------------------
template <int JT>
__global__ __launch_bounds__(256)
void k_gemm(const bf16* __restrict__ A, long long sA,
            const bf16* __restrict__ B, long long sB,
            float* __restrict__ C, long long sC,
            const float* __restrict__ bias, long long sBias,
            int M, int N, int K, int lda, int ldb, int ldc)
{
    int bz = blockIdx.z;
    A += bz * sA; B += bz * sB; C += bz * sC;
    if (bias) bias += bz * sBias;

    int lane = threadIdx.x & 31;
    int wave = threadIdx.x >> 5;
    int wm = wave >> 1, wn = wave & 1;
    int m0 = blockIdx.y * 128 + wm * 32;
    int n0 = blockIdx.x * (32 * JT) + wn * (16 * JT);

    int row16 = lane & 15;
    int kbA = (lane >> 4) * 8;     // A half-row select
    int kbB = (lane >> 4) * 16;    // unused (B loads use kbA)

    int ar[2], br[JT];
#pragma unroll
    for (int i = 0; i < 2; i++)  ar[i] = imin(m0 + i * 16 + row16, M - 1);
#pragma unroll
    for (int j = 0; j < JT; j++) br[j] = imin(n0 + j * 16 + row16, N - 1);

    v8f acc[2][JT] = {};
    for (int k0 = 0; k0 < K; k0 += 32) {
        if (k0 + 32 < K) {
            __builtin_prefetch(A + (long long)ar[0] * lda + k0 + 32, 0, 1);
            __builtin_prefetch(B + (long long)br[0] * ldb + k0 + 32, 0, 1);
        }
        v16bf a[2], b[JT];
#pragma unroll
        for (int i = 0; i < 2; i++) {
            const bf16* pa = A + (long long)ar[i] * lda + k0 + kbA;
            v8bf lo = *(const v8bf*)pa;
            v8bf hi = *(const v8bf*)(pa + 16);
#pragma unroll
            for (int e = 0; e < 8; e++) { a[i][e] = lo[e]; a[i][8 + e] = hi[e]; }
        }
#pragma unroll
        for (int j = 0; j < JT; j++) {
            const bf16* pb = B + (long long)br[j] * ldb + k0 + kbA;
            v8bf lo = *(const v8bf*)pb;
            v8bf hi = *(const v8bf*)(pb + 16);
#pragma unroll
            for (int e = 0; e < 8; e++) { b[j][e] = lo[e]; b[j][8 + e] = hi[e]; }
        }
#pragma unroll
        for (int i = 0; i < 2; i++)
#pragma unroll
            for (int j = 0; j < JT; j++)
                acc[i][j] = __builtin_amdgcn_wmma_f32_16x16x32_bf16(
                    false, a[i], false, b[j], (short)0, acc[i][j], false, false);
    }

    int col16 = lane & 15;
    int rbase = (lane >> 4) * 8;
#pragma unroll
    for (int i = 0; i < 2; i++) {
#pragma unroll
        for (int j = 0; j < JT; j++) {
            int ccol = n0 + j * 16 + col16;
            if (ccol >= N) continue;
            float bv = bias ? bias[ccol] : 0.0f;
#pragma unroll
            for (int e = 0; e < 8; e++) {
                int crow = m0 + i * 16 + rbase + e;
                if (crow < M)
                    C[(long long)crow * ldc + ccol] = acc[i][j][e] + bv;
            }
        }
    }
}

// f32 [R,C] (batch stride sIn) -> bf16 transpose out[c*ldout + r] (batch stride sOut)
__global__ __launch_bounds__(256)
void k_transpose_bf16(const float* __restrict__ in, long long sIn,
                      bf16* __restrict__ outp, long long sOut,
                      int R, int C, int ldout)
{
    __shared__ float tile[32][33];
    in += (long long)blockIdx.z * sIn;
    outp += (long long)blockIdx.z * sOut;
    int r0 = blockIdx.y * 32, c0 = blockIdx.x * 32;
    int tx = threadIdx.x & 31, ty = threadIdx.x >> 5;
    for (int rr = ty; rr < 32; rr += 8) {
        int r = r0 + rr, c = c0 + tx;
        tile[rr][tx] = (r < R && c < C) ? in[(long long)r * C + c] : 0.0f;
    }
    __syncthreads();
    for (int rr = ty; rr < 32; rr += 8) {
        int c = c0 + rr, r = r0 + tx;
        if (c < C && r < R) outp[(long long)c * ldout + r] = f2bf(tile[tx][rr]);
    }
}

__global__ void k_convert_bf16(const float* __restrict__ in, bf16* __restrict__ outp, long long n) {
    long long i = (long long)blockIdx.x * blockDim.x + threadIdx.x;
    long long st = (long long)gridDim.x * blockDim.x;
    for (; i < n; i += st) outp[i] = f2bf(in[i]);
}

// LayerNorm over D per row; optional fused SiLU (act==1); optional f32/bf16 outputs.
__global__ __launch_bounds__(256)
void k_layernorm(const float* __restrict__ x, const float* __restrict__ g,
                 const float* __restrict__ b, float* __restrict__ yf,
                 bf16* __restrict__ yb, int D, int act)
{
    long long row = blockIdx.x;
    const float* xr = x + row * D;
    int tid = threadIdx.x;
    __shared__ float s1[256], s2[256];
    float a0 = 0.f, a1 = 0.f;
    for (int j = tid; j < D; j += 256) { float v = xr[j]; a0 += v; a1 += v * v; }
    s1[tid] = a0; s2[tid] = a1; __syncthreads();
    for (int o = 128; o > 0; o >>= 1) {
        if (tid < o) { s1[tid] += s1[tid + o]; s2[tid] += s2[tid + o]; }
        __syncthreads();
    }
    float mean = s1[0] / (float)D;
    float var  = s2[0] / (float)D - mean * mean;
    float rst  = rsqrtf(var + 1e-5f);
    for (int j = tid; j < D; j += 256) {
        float v = (xr[j] - mean) * rst * g[j] + b[j];
        if (act == 1) v = v * (1.0f / (1.0f + __expf(-v)));      // SiLU
        if (yf) yf[row * D + j] = v;
        if (yb) yb[row * D + j] = f2bf(v);
    }
}

__device__ __forceinline__ float logit_at(const float* __restrict__ ac,
                                          const float* __restrict__ bd,
                                          int t, int s, int S, float scale)
{
    float l = ac[(long long)t * S + s];
    if (bd) {  // Transformer-XL rel-shift gather
        long long f = (long long)(t + 1) * S + s;
        long long t2 = f / (long long)(S + 1);
        int jj = (int)(f % (long long)(S + 1));
        l += (jj == 0) ? 0.0f : bd[t2 * S + (jj - 1)];
    }
    return l * scale;
}

// Row softmax (with optional fused rel-shift), bf16 probs out. One block per row.
__global__ __launch_bounds__(256)
void k_softmax(const float* __restrict__ ac, const float* __restrict__ bd,
               bf16* __restrict__ pr, float scale, int S)
{
    int t = blockIdx.x;
    bf16* prow = pr + (long long)t * S;
    int tid = threadIdx.x;
    __shared__ float red[256];
    float mx = -3.0e38f;
    for (int s = tid; s < S; s += 256) mx = fmaxf(mx, logit_at(ac, bd, t, s, S, scale));
    red[tid] = mx; __syncthreads();
    for (int o = 128; o > 0; o >>= 1) { if (tid < o) red[tid] = fmaxf(red[tid], red[tid + o]); __syncthreads(); }
    mx = red[0]; __syncthreads();
    float sum = 0.f;
    for (int s = tid; s < S; s += 256) sum += __expf(logit_at(ac, bd, t, s, S, scale) - mx);
    red[tid] = sum; __syncthreads();
    for (int o = 128; o > 0; o >>= 1) { if (tid < o) red[tid] += red[tid + o]; __syncthreads(); }
    float inv = 1.0f / red[0];
    for (int s = tid; s < S; s += 256)
        prow[s] = f2bf(__expf(logit_at(ac, bd, t, s, S, scale) - mx) * inv);
}

// out_bf16[h,t,k] = q[h,t,k] + bias[h,k]
__global__ void k_headbias(const float* __restrict__ q, const float* __restrict__ bias,
                           bf16* __restrict__ outp, long long n, int TDH, int DH)
{
    long long i = (long long)blockIdx.x * blockDim.x + threadIdx.x;
    long long st = (long long)gridDim.x * blockDim.x;
    for (; i < n; i += st) {
        int hh = (int)(i / TDH);
        int kk = (int)(i % DH);
        outp[i] = f2bf(q[i] + bias[hh * DH + kk]);
    }
}

__global__ void k_glu(const float* __restrict__ in, float* __restrict__ outp, int T, int D) {
    long long i = (long long)blockIdx.x * blockDim.x + threadIdx.x;
    long long n = (long long)T * D;
    long long st = (long long)gridDim.x * blockDim.x;
    for (; i < n; i += st) {
        long long t = i / D, d = i % D;
        float a = in[t * 2 * D + d];
        float g = in[t * 2 * D + D + d];
        outp[i] = a * (1.0f / (1.0f + __expf(-g)));
    }
}

__global__ void k_dwconv(const float* __restrict__ in, const float* __restrict__ w,
                         const float* __restrict__ bias, float* __restrict__ outp,
                         int T, int D, int K)
{
    long long i = (long long)blockIdx.x * blockDim.x + threadIdx.x;
    if (i >= (long long)T * D) return;
    int t = (int)(i / D), d = (int)(i % D);
    int P = K / 2;
    float acc = bias[d];
    for (int j = 0; j < K; j++) {
        int tt = t + j - P;
        if (tt >= 0 && tt < T) acc += in[(long long)tt * D + d] * w[d * K + j];
    }
    outp[i] = acc;
}

__global__ void k_silu_bf16(const float* __restrict__ in, bf16* __restrict__ outp, long long n) {
    long long i = (long long)blockIdx.x * blockDim.x + threadIdx.x;
    long long st = (long long)gridDim.x * blockDim.x;
    for (; i < n; i += st) {
        float v = in[i];
        outp[i] = f2bf(v * (1.0f / (1.0f + __expf(-v))));
    }
}

__global__ void k_gelu_bf16(const float* __restrict__ in, bf16* __restrict__ outp, long long n) {
    long long i = (long long)blockIdx.x * blockDim.x + threadIdx.x;
    long long st = (long long)gridDim.x * blockDim.x;
    for (; i < n; i += st) {
        float v = in[i];
        float c = 0.7978845608028654f * (v + 0.044715f * v * v * v);
        outp[i] = f2bf(0.5f * v * (1.0f + tanhf(c)));
    }
}

__global__ void k_add(float* __restrict__ dst, const float* __restrict__ src, long long n) {
    long long i = (long long)blockIdx.x * blockDim.x + threadIdx.x;
    long long st = (long long)gridDim.x * blockDim.x;
    for (; i < n; i += st) dst[i] += src[i];
}

__global__ void k_copy(float* __restrict__ dst, const float* __restrict__ src, long long n) {
    long long i = (long long)blockIdx.x * blockDim.x + threadIdx.x;
    long long st = (long long)gridDim.x * blockDim.x;
    for (; i < n; i += st) dst[i] = src[i];
}

// per-column (D blocks) mean / std over T rows (std = sqrt(clip(E[x^2]-mean^2,1e-6)))
__global__ __launch_bounds__(256)
void k_colstats(const float* __restrict__ h, float* __restrict__ meanp,
                float* __restrict__ stdp, int T, int D)
{
    int d = blockIdx.x;
    int tid = threadIdx.x;
    __shared__ float s1[256], s2[256];
    float a0 = 0.f, a1 = 0.f;
    for (int t = tid; t < T; t += 256) {
        float v = h[(long long)t * D + d];
        a0 += v; a1 += v * v;
    }
    s1[tid] = a0; s2[tid] = a1; __syncthreads();
    for (int o = 128; o > 0; o >>= 1) {
        if (tid < o) { s1[tid] += s1[tid + o]; s2[tid] += s2[tid + o]; }
        __syncthreads();
    }
    if (tid == 0) {
        float m = s1[0] / (float)T;
        float var = s2[0] / (float)T - m * m;
        meanp[d] = m;
        stdp[d] = sqrtf(fmaxf(var, 1e-6f));
    }
}

// ---------------------------------------------------------------------------
// Host orchestration
// ---------------------------------------------------------------------------
extern "C" void kernel_launch(void* const* d_in, const int* in_sizes, int n_in,
                              void* d_out, int out_size, void* d_ws, size_t ws_size,
                              hipStream_t stream)
{
    (void)in_sizes; (void)n_in; (void)out_size; (void)ws_size;

    const float* x      = (const float*)d_in[0];
    const float* pos    = (const float*)d_in[1];
    const float* lat0   = (const float*)d_in[2];
    const float* enclns = (const float*)d_in[3];
    const float* Wq     = (const float*)d_in[4];
    const float* Wk     = (const float*)d_in[5];
    const float* Wv     = (const float*)d_in[6];
    const float* Wpos   = (const float*)d_in[7];
    const float* bqkv   = (const float*)d_in[8];
    const float* biasuv = (const float*)d_in[9];
    const float* Wout   = (const float*)d_in[10];
    const float* bout   = (const float*)d_in[11];
    const float* pw1w   = (const float*)d_in[12];
    const float* pw1b   = (const float*)d_in[13];
    const float* dww    = (const float*)d_in[14];
    const float* dwb    = (const float*)d_in[15];
    const float* pw2w   = (const float*)d_in[16];
    const float* pw2b   = (const float*)d_in[17];
    const float* ffw1   = (const float*)d_in[18];
    const float* ffb1   = (const float*)d_in[19];
    const float* ffw2   = (const float*)d_in[20];
    const float* ffb2   = (const float*)d_in[21];
    const float* afterln= (const float*)d_in[22];
    const float* projw  = (const float*)d_in[23];
    const float* projb  = (const float*)d_in[24];
    const float* plns   = (const float*)d_in[25];
    const float* pWqw   = (const float*)d_in[26];
    const float* pWkw   = (const float*)d_in[27];
    const float* pWvw   = (const float*)d_in[28];
    const float* pWoutw = (const float*)d_in[29];
    const float* pffw1  = (const float*)d_in[30];
    const float* pffb1  = (const float*)d_in[31];
    const float* pffw2  = (const float*)d_in[32];
    const float* pffb2  = (const float*)d_in[33];
    const float* finln  = (const float*)d_in[34];
    float* out = (float*)d_out;

    // --- workspace bump arena ---
    char* base = (char*)d_ws;
    size_t off = 0;
    auto af = [&](size_t e) -> float* { off = (off + 255) & ~(size_t)255; float* p = (float*)(base + off); off += e * 4; return p; };
    auto ab = [&](size_t e) -> bf16*  { off = (off + 255) & ~(size_t)255; bf16*  p = (bf16*)(base + off);  off += e * 2; return p; };

    float* hbuf  = af((size_t)T_ * D_);
    float* qf    = af((size_t)H_ * T_ * DH_);
    float* kf    = af((size_t)H_ * T_ * DH_);
    float* vf    = af((size_t)H_ * T_ * DH_);
    float* pf    = af((size_t)H_ * T_ * DH_);
    float* acb   = af((size_t)T_ * T_);
    float* bdb   = af((size_t)T_ * T_);
    float* aoc   = af((size_t)T_ * H_ * DH_);
    float* tbig  = af((size_t)T_ * FF_);
    float* tDa   = af((size_t)T_ * D_);
    float* tDb   = af((size_t)T_ * D_);
    float* latb  = af((size_t)NLAT_ * D_);
    float* lattmp= af((size_t)NLAT_ * D_);
    float* pqf   = af((size_t)PH_ * NLAT_ * PDH_);
    float* pkk   = af((size_t)PH_ * S_ * PDH_);
    float* pvv   = af((size_t)PH_ * S_ * PDH_);
    float* pscore= af((size_t)PH_ * NLAT_ * S_);
    float* paoc  = af((size_t)NLAT_ * PH_ * PDH_);
    float* pff1  = af((size_t)NLAT_ * 4 * D_);
    float* meanb = af((size_t)D_);
    float* stdb  = af((size_t)D_);

    bf16* pe_bf   = ab((size_t)T_ * D_);
    bf16* xln_bf  = ab((size_t)T_ * D_);
    bf16* qu_bf   = ab((size_t)H_ * T_ * DH_);
    bf16* qv_bf   = ab((size_t)H_ * T_ * DH_);
    bf16* kbf     = ab((size_t)H_ * T_ * DH_);
    bf16* pbf     = ab((size_t)H_ * T_ * DH_);
    bf16* vt_bf   = ab((size_t)H_ * DH_ * T_);
    bf16* probs   = ab((size_t)H_ * T_ * T_);
    bf16* aoc_bf  = ab((size_t)T_ * H_ * DH_);
    bf16* tbig_bf = ab((size_t)T_ * FF_);
    bf16* kvin_bf = ab((size_t)S_ * D_);
    bf16* pq_bf   = ab((size_t)PH_ * NLAT_ * PDH_);
    bf16* pkk_bf  = ab((size_t)PH_ * S_ * PDH_);
    bf16* pvt_bf  = ab((size_t)PH_ * PDH_ * S_);
    bf16* pprob   = ab((size_t)PH_ * NLAT_ * S_);
    bf16* paoc_bf = ab((size_t)NLAT_ * PH_ * PDH_);
    bf16* latln_bf= ab((size_t)NLAT_ * D_);
    bf16* pff1_bf = ab((size_t)NLAT_ * 4 * D_);
    // weight staging (bf16, reused per layer)
    bf16* wq_t   = ab((size_t)H_ * DH_ * D_);
    bf16* wk_t   = ab((size_t)H_ * DH_ * D_);
    bf16* wv_t   = ab((size_t)H_ * DH_ * D_);
    bf16* wpos_t = ab((size_t)H_ * DH_ * D_);
    bf16* wout_t = ab((size_t)D_ * H_ * DH_);
    bf16* pw1_bf = ab((size_t)2 * D_ * D_);
    bf16* pw2_bf = ab((size_t)D_ * D_);
    bf16* ffw1_t = ab((size_t)FF_ * D_);
    bf16* ffw2_t = ab((size_t)D_ * FF_);
    bf16* projw_t= ab((size_t)D_ * D_);
    bf16* pwq_t  = ab((size_t)PH_ * PDH_ * D_);
    bf16* pwk_t  = ab((size_t)PH_ * PDH_ * D_);
    bf16* pwv_t  = ab((size_t)PH_ * PDH_ * D_);
    bf16* pwout_t= ab((size_t)D_ * PH_ * PDH_);
    bf16* pffw1_t= ab((size_t)4 * D_ * D_);
    bf16* pffw2_t= ab((size_t)D_ * 4 * D_);

    // --- launch helpers ---
    auto ewg = [](long long n) -> dim3 {
        long long b = (n + 255) / 256; if (b > 16384) b = 16384; if (b < 1) b = 1;
        return dim3((unsigned)b);
    };
    auto gemm = [&](const bf16* A, long long sA, int lda,
                    const bf16* B, long long sB, int ldb,
                    float* C, long long sC, int ldc,
                    const float* bias, long long sBias,
                    int M, int N, int K, int batch) {
        if (N >= 128) {  // wide tiles: wave 32x64, 8 WMMA per k-step
            dim3 grid((N + 127) / 128, (M + 127) / 128, batch);
            k_gemm<4><<<grid, 256, 0, stream>>>(A, sA, B, sB, C, sC, bias, sBias, M, N, K, lda, ldb, ldc);
        } else {         // narrow tiles for N==64 projections
            dim3 grid((N + 63) / 64, (M + 127) / 128, batch);
            k_gemm<2><<<grid, 256, 0, stream>>>(A, sA, B, sB, C, sC, bias, sBias, M, N, K, lda, ldb, ldc);
        }
    };
    auto transp = [&](const float* in, long long sIn, int R, int C,
                      bf16* o, long long sOut, int ldout, int batch) {
        dim3 grid((C + 31) / 32, (R + 31) / 32, batch);
        k_transpose_bf16<<<grid, 256, 0, stream>>>(in, sIn, o, sOut, R, C, ldout);
    };
    auto cvt = [&](const float* in, bf16* o, long long n) {
        k_convert_bf16<<<ewg(n), 256, 0, stream>>>(in, o, n);
    };
    auto lnorm = [&](const float* xx, const float* gb, float* yf, bf16* yb, int rows, int act) {
        k_layernorm<<<rows, 256, 0, stream>>>(xx, gb, gb + D_, yf, yb, D_, act);
    };
    auto add = [&](float* d, const float* s, long long n) { k_add<<<ewg(n), 256, 0, stream>>>(d, s, n); };
    auto cpy = [&](float* d, const float* s, long long n) { k_copy<<<ewg(n), 256, 0, stream>>>(d, s, n); };

    // --- init: h = x, pe = bf16(pos_emb) ---
    cpy(hbuf, x, (long long)T_ * D_);
    cvt(pos, pe_bf, (long long)T_ * D_);

    const float scale = 0.125f;  // DH^-0.5

    // =========================== encoder layers ===========================
    for (int i = 0; i < NL_; i++) {
        const float* ln_i = enclns + (size_t)i * 5 * 2 * D_;
        // weight prep -> bf16, K-contiguous
        transp(Wq   + (size_t)i * H_ * D_ * DH_, (long long)D_ * DH_, D_, DH_, wq_t,   (long long)DH_ * D_, D_, H_);
        transp(Wk   + (size_t)i * H_ * D_ * DH_, (long long)D_ * DH_, D_, DH_, wk_t,   (long long)DH_ * D_, D_, H_);
        transp(Wv   + (size_t)i * H_ * D_ * DH_, (long long)D_ * DH_, D_, DH_, wv_t,   (long long)DH_ * D_, D_, H_);
        transp(Wpos + (size_t)i * H_ * D_ * DH_, (long long)D_ * DH_, D_, DH_, wpos_t, (long long)DH_ * D_, D_, H_);
        transp(Wout + (size_t)i * H_ * DH_ * D_, (long long)DH_ * D_, DH_, D_, wout_t, (long long)DH_, H_ * DH_, H_);
        cvt(pw1w + (size_t)i * 2 * D_ * D_, pw1_bf, (long long)2 * D_ * D_);
        cvt(pw2w + (size_t)i * D_ * D_,     pw2_bf, (long long)D_ * D_);
        transp(ffw1 + (size_t)i * D_ * FF_, 0, D_, FF_, ffw1_t, 0, D_, 1);
        transp(ffw2 + (size_t)i * FF_ * D_, 0, FF_, D_, ffw2_t, 0, FF_, 1);

        // --- rel-pos MHA ---
        lnorm(hbuf, ln_i + 0 * 2 * D_, nullptr, xln_bf, T_, 0);
        gemm(xln_bf, 0, D_, wq_t,   (long long)DH_ * D_, D_, qf, (long long)T_ * DH_, DH_,
             bqkv + ((size_t)i * 3 + 0) * H_ * DH_, DH_, T_, DH_, D_, H_);
        gemm(xln_bf, 0, D_, wk_t,   (long long)DH_ * D_, D_, kf, (long long)T_ * DH_, DH_,
             bqkv + ((size_t)i * 3 + 1) * H_ * DH_, DH_, T_, DH_, D_, H_);
        gemm(xln_bf, 0, D_, wv_t,   (long long)DH_ * D_, D_, vf, (long long)T_ * DH_, DH_,
             bqkv + ((size_t)i * 3 + 2) * H_ * DH_, DH_, T_, DH_, D_, H_);
        gemm(pe_bf,  0, D_, wpos_t, (long long)DH_ * D_, D_, pf, (long long)T_ * DH_, DH_,
             nullptr, 0, T_, DH_, D_, H_);

        long long nq = (long long)H_ * T_ * DH_;
        k_headbias<<<ewg(nq), 256, 0, stream>>>(qf, biasuv + ((size_t)i * 2 + 0) * H_ * DH_, qu_bf, nq, T_ * DH_, DH_);
        k_headbias<<<ewg(nq), 256, 0, stream>>>(qf, biasuv + ((size_t)i * 2 + 1) * H_ * DH_, qv_bf, nq, T_ * DH_, DH_);
        cvt(kf, kbf, nq);
        cvt(pf, pbf, nq);
        transp(vf, (long long)T_ * DH_, T_, DH_, vt_bf, (long long)DH_ * T_, T_, H_);

        for (int hh = 0; hh < H_; hh++) {
            gemm(qu_bf + (size_t)hh * T_ * DH_, 0, DH_, kbf + (size_t)hh * T_ * DH_, 0, DH_,
                 acb, 0, T_, nullptr, 0, T_, T_, DH_, 1);
            gemm(qv_bf + (size_t)hh * T_ * DH_, 0, DH_, pbf + (size_t)hh * T_ * DH_, 0, DH_,
                 bdb, 0, T_, nullptr, 0, T_, T_, DH_, 1);
            k_softmax<<<T_, 256, 0, stream>>>(acb, bdb, probs + (size_t)hh * T_ * T_, scale, T_);
        }
        gemm(probs, (long long)T_ * T_, T_, vt_bf, (long long)DH_ * T_, T_,
             aoc, (long long)DH_, H_ * DH_, nullptr, 0, T_, DH_, T_, H_);
        cvt(aoc, aoc_bf, (long long)T_ * H_ * DH_);
        gemm(aoc_bf, 0, H_ * DH_, wout_t, 0, H_ * DH_, tDa, 0, D_,
             bout + (size_t)i * D_, 0, T_, D_, H_ * DH_, 1);
        add(hbuf, tDa, (long long)T_ * D_);

        // --- conformer conv module ---
        lnorm(hbuf, ln_i + 1 * 2 * D_, nullptr, xln_bf, T_, 0);
        gemm(xln_bf, 0, D_, pw1_bf, 0, D_, tbig, 0, 2 * D_,
             pw1b + (size_t)i * 2 * D_, 0, T_, 2 * D_, D_, 1);
        k_glu<<<ewg((long long)T_ * D_), 256, 0, stream>>>(tbig, tDa, T_, D_);
        k_dwconv<<<dim3((T_ * D_ + 255) / 256), 256, 0, stream>>>(
            tDa, dww + (size_t)i * D_ * KC_, dwb + (size_t)i * D_, tDb, T_, D_, KC_);
        lnorm(tDb, ln_i + 2 * 2 * D_, nullptr, xln_bf, T_, 1 /*silu*/);
        gemm(xln_bf, 0, D_, pw2_bf, 0, D_, tDa, 0, D_,
             pw2b + (size_t)i * D_, 0, T_, D_, D_, 1);
        add(hbuf, tDa, (long long)T_ * D_);

        // --- feed-forward ---
        lnorm(hbuf, ln_i + 3 * 2 * D_, nullptr, xln_bf, T_, 0);
        gemm(xln_bf, 0, D_, ffw1_t, 0, D_, tbig, 0, FF_,
             ffb1 + (size_t)i * FF_, 0, T_, FF_, D_, 1);
        k_silu_bf16<<<ewg((long long)T_ * FF_), 256, 0, stream>>>(tbig, tbig_bf, (long long)T_ * FF_);
        gemm(tbig_bf, 0, FF_, ffw2_t, 0, FF_, tDa, 0, D_,
             ffb2 + (size_t)i * D_, 0, T_, D_, FF_, 1);
        add(hbuf, tDa, (long long)T_ * D_);

        // --- layer final LN (in-place) ---
        lnorm(hbuf, ln_i + 4 * 2 * D_, hbuf, nullptr, T_, 0);
    }

    // ======================= after-ln + projection =======================
    lnorm(hbuf, afterln, nullptr, xln_bf, T_, 0);
    transp(projw, 0, D_, D_, projw_t, 0, D_, 1);
    gemm(xln_bf, 0, D_, projw_t, 0, D_, tDa, 0, D_, projb, 0, T_, D_, D_, 1);
    cpy(hbuf, tDa, (long long)T_ * D_);
    k_colstats<<<D_, 256, 0, stream>>>(hbuf, meanb, stdb, T_, D_);

    // ========================= perceiver resampler =========================
    cpy(latb, lat0, (long long)NLAT_ * D_);
    for (int j = 0; j < NP_; j++) {
        const float* pl = plns + (size_t)j * 3 * 2 * D_;
        transp(pWqw + (size_t)j * PH_ * D_ * PDH_, (long long)D_ * PDH_, D_, PDH_, pwq_t, (long long)PDH_ * D_, D_, PH_);
        transp(pWkw + (size_t)j * PH_ * D_ * PDH_, (long long)D_ * PDH_, D_, PDH_, pwk_t, (long long)PDH_ * D_, D_, PH_);
        transp(pWvw + (size_t)j * PH_ * D_ * PDH_, (long long)D_ * PDH_, D_, PDH_, pwv_t, (long long)PDH_ * D_, D_, PH_);
        transp(pWoutw + (size_t)j * PH_ * PDH_ * D_, (long long)PDH_ * D_, PDH_, D_, pwout_t, (long long)PDH_, PH_ * PDH_, PH_);
        transp(pffw1 + (size_t)j * D_ * 4 * D_, 0, D_, 4 * D_, pffw1_t, 0, D_, 1);
        transp(pffw2 + (size_t)j * 4 * D_ * D_, 0, 4 * D_, D_, pffw2_t, 0, 4 * D_, 1);

        // kvin = [LN(h) ; LN(lat)] in bf16
        lnorm(latb, pl + 0 * 2 * D_, nullptr, kvin_bf + (size_t)T_ * D_, NLAT_, 0);
        lnorm(hbuf, pl + 1 * 2 * D_, nullptr, kvin_bf, T_, 0);

        gemm(kvin_bf + (size_t)T_ * D_, 0, D_, pwq_t, (long long)PDH_ * D_, D_,
             pqf, (long long)NLAT_ * PDH_, PDH_, nullptr, 0, NLAT_, PDH_, D_, PH_);
        gemm(kvin_bf, 0, D_, pwk_t, (long long)PDH_ * D_, D_,
             pkk, (long long)S_ * PDH_, PDH_, nullptr, 0, S_, PDH_, D_, PH_);
        gemm(kvin_bf, 0, D_, pwv_t, (long long)PDH_ * D_, D_,
             pvv, (long long)S_ * PDH_, PDH_, nullptr, 0, S_, PDH_, D_, PH_);
        cvt(pqf, pq_bf, (long long)PH_ * NLAT_ * PDH_);
        cvt(pkk, pkk_bf, (long long)PH_ * S_ * PDH_);
        transp(pvv, (long long)S_ * PDH_, S_, PDH_, pvt_bf, (long long)PDH_ * S_, S_, PH_);

        gemm(pq_bf, (long long)NLAT_ * PDH_, PDH_, pkk_bf, (long long)S_ * PDH_, PDH_,
             pscore, (long long)NLAT_ * S_, S_, nullptr, 0, NLAT_, S_, PDH_, PH_);
        k_softmax<<<PH_ * NLAT_, 256, 0, stream>>>(pscore, nullptr, pprob, 0.125f, S_);
        gemm(pprob, (long long)NLAT_ * S_, S_, pvt_bf, (long long)PDH_ * S_, S_,
             paoc, (long long)PDH_, PH_ * PDH_, nullptr, 0, NLAT_, PDH_, S_, PH_);
        cvt(paoc, paoc_bf, (long long)NLAT_ * PH_ * PDH_);
        gemm(paoc_bf, 0, PH_ * PDH_, pwout_t, 0, PH_ * PDH_, lattmp, 0, D_,
             nullptr, 0, NLAT_, D_, PH_ * PDH_, 1);
        add(latb, lattmp, (long long)NLAT_ * D_);

        lnorm(latb, pl + 2 * 2 * D_, nullptr, latln_bf, NLAT_, 0);
        gemm(latln_bf, 0, D_, pffw1_t, 0, D_, pff1, 0, 4 * D_,
             pffb1 + (size_t)j * 4 * D_, 0, NLAT_, 4 * D_, D_, 1);
        k_gelu_bf16<<<ewg((long long)NLAT_ * 4 * D_), 256, 0, stream>>>(pff1, pff1_bf, (long long)NLAT_ * 4 * D_);
        gemm(pff1_bf, 0, 4 * D_, pffw2_t, 0, 4 * D_, lattmp, 0, D_,
             pffb2 + (size_t)j * D_, 0, NLAT_, D_, 4 * D_, 1);
        add(latb, lattmp, (long long)NLAT_ * D_);
    }

    // final LN of latents straight into d_out rows [0,32); then mean/std rows
    lnorm(latb, finln, out, nullptr, NLAT_, 0);
    cpy(out + (size_t)NLAT_ * D_,       meanb, D_);
    cpy(out + (size_t)(NLAT_ + 1) * D_, stdb,  D_);
}
